// LogSparseAttention_66632122630794
// MI455X (gfx1250) — matrix operand order repaired and checked by reading, not compiled, not run on hardware
//
#include <hip/hip_runtime.h>
#include <hip/hip_bf16.h>

#define B_  2
#define L_  2048
#define D_  1024
#define H_  16
#define DH_ 64

typedef __bf16 v16bf __attribute__((ext_vector_type(16)));
typedef __bf16 v8bf  __attribute__((ext_vector_type(8)));
typedef float  v8f   __attribute__((ext_vector_type(8)));

__device__ __forceinline__ unsigned short f32_bf16_rne(float f) {
    unsigned int u = __float_as_uint(f);
    u += 0x7FFFu + ((u >> 16) & 1u);          // round-to-nearest-even
    return (unsigned short)(u >> 16);
}

// Low 32 bits of a generic pointer into LDS == LDS byte offset (ISA 10.2 aperture).
__device__ __forceinline__ unsigned lds_off(const void* p) {
    return (unsigned)(size_t)p;
}

// CDNA5 async copy global->LDS, tracked by ASYNCcnt (ISA 10 / 15.18.3 op 98).
#define ASYNC_LOAD_B128(ldsoff, gptr)                                          \
    asm volatile("global_load_async_to_lds_b128 %0, %1, off"                   \
                 :: "v"(ldsoff), "v"(gptr) : "memory")
#define WAIT_ASYNCCNT_8() asm volatile("s_wait_asynccnt 8" ::: "memory")
#define WAIT_ASYNCCNT_0() asm volatile("s_wait_asynccnt 0" ::: "memory")

// ---------------------------------------------------------------------------
// Stage 1: fp32 -> bf16 cast (vectorized 4 elements / thread)
// ---------------------------------------------------------------------------
__global__ void cast_bf16_kernel(const float* __restrict__ src,
                                 unsigned short* __restrict__ dst, int n4) {
    int i = blockIdx.x * blockDim.x + threadIdx.x;
    if (i >= n4) return;
    float4 f = reinterpret_cast<const float4*>(src)[i];
    uint2 p;
    p.x = (unsigned int)f32_bf16_rne(f.x) | ((unsigned int)f32_bf16_rne(f.y) << 16);
    p.y = (unsigned int)f32_bf16_rne(f.z) | ((unsigned int)f32_bf16_rne(f.w) << 16);
    reinterpret_cast<uint2*>(dst)[i] = p;
}

// ---------------------------------------------------------------------------
// Stage 2/5: out[M,N] = A[M,K](bf16,rm) @ W[N,K](bf16,rm)^T + bias, fp32 out.
//
// Block: 256 thr (8 waves, grid 4(M) x 2(N)). Block tile 128x128.
// K chunk = 64, double-buffered in LDS via global_load_async_to_lds_b128:
//   As[buf][128][64], Bs[buf][128][64] (16 KB each, 64 KB total).
// Each wave: 32(M) x 64(N) = 8 accumulators. Per 32-wide K substep all 12
// fragment ds_load_b128 (2 A frags + 4 B frags, distinct registers) are
// issued before the 8 v_wmma_f32_16x16x32_bf16, so DScnt drains gradually
// instead of a hard s_wait_dscnt 0 per WMMA pair.
// Fragment lane layouts per ISA 7.12.2 (A: 16x32 bf16; B: 32x16 col-contig;
// C/D: VGPR r -> M = r + 8*(lane>=16), N = lane&15).
// ---------------------------------------------------------------------------
__global__ __launch_bounds__(256)
void gemm_bf16_wmma(const __bf16* __restrict__ A, const __bf16* __restrict__ W,
                    const float* __restrict__ bias, float* __restrict__ out,
                    int M, int N, int K) {
    __shared__ __align__(16) unsigned short As[2][128][64];
    __shared__ __align__(16) unsigned short Bs[2][128][64];

    const int tid   = threadIdx.x;
    const int lane  = tid & 31;
    const int wave  = tid >> 5;
    const int half  = lane >> 4;
    const int l15   = lane & 15;
    const int waveM = wave & 3;      // 4 waves along M (32 rows each)
    const int waveN = wave >> 2;     // 2 waves along N (64 cols each)

    const __bf16* Ablk = A + (size_t)blockIdx.x * 128 * K;
    const __bf16* Wblk = W + (size_t)blockIdx.y * 128 * K;

    v8f c[2][4] = {};

    // Stage one 128x64 A tile + 128x64 B tile: 1024 16B chunks each,
    // 256 threads x 4 rounds x (1 A + 1 B) = 8 async issues per thread.
    auto stage = [&](int kc, int sbuf) {
        const int k0 = kc * 64;
#pragma unroll
        for (int r = 0; r < 4; ++r) {
            const int cix = tid + 256 * r;      // 0..1023
            const int row = cix >> 3;           // 0..127
            const int ko  = (cix & 7) * 8;      // 0,8,...,56
            const __bf16* ga = Ablk + (size_t)row * K + k0 + ko;
            const __bf16* gb = Wblk + (size_t)row * K + k0 + ko;
            const unsigned la = lds_off(&As[sbuf][row][ko]);
            const unsigned lb = lds_off(&Bs[sbuf][row][ko]);
            ASYNC_LOAD_B128(la, ga);
            ASYNC_LOAD_B128(lb, gb);
        }
    };

    const int NK = K / 64;
    stage(0, 0);
    int buf = 0;
    for (int kc = 0; kc < NK; ++kc) {
        if (kc + 1 < NK) {
            stage(kc + 1, buf ^ 1);   // prefetch next chunk into other buffer
            WAIT_ASYNCCNT_8();        // my 8 issues for chunk kc are complete
        } else {
            WAIT_ASYNCCNT_0();
        }
        __syncthreads();              // all waves' async writes visible

#pragma unroll
        for (int ks = 0; ks < 64; ks += 32) {
            union { v16bf v; v8bf h[2]; } a[2], b[4];
            // Issue ALL fragment loads first (distinct registers) ...
#pragma unroll
            for (int mt = 0; mt < 2; ++mt) {
                const unsigned short* ap = &As[buf][waveM * 32 + mt * 16 + l15][0];
                a[mt].h[0] = *reinterpret_cast<const v8bf*>(ap + ks + half * 8);
                a[mt].h[1] = *reinterpret_cast<const v8bf*>(ap + ks + 16 + half * 8);
            }
#pragma unroll
            for (int t = 0; t < 4; ++t) {
                const unsigned short* bp =
                    &Bs[buf][waveN * 64 + t * 16 + l15][0] + ks + half * 8;
                b[t].h[0] = *reinterpret_cast<const v8bf*>(bp);
                b[t].h[1] = *reinterpret_cast<const v8bf*>(bp + 16);
            }
            // ... then stream the 8 WMMAs.
#pragma unroll
            for (int t = 0; t < 4; ++t)
#pragma unroll
                for (int mt = 0; mt < 2; ++mt)
                    c[mt][t] = __builtin_amdgcn_wmma_f32_16x16x32_bf16(
                                   false, a[mt].v, false, b[t].v, (short)0,
                                   c[mt][t], false, false);
        }
        __syncthreads();              // done reading buf before it is refilled
        buf ^= 1;
    }

#pragma unroll
    for (int t = 0; t < 4; ++t) {
        const int n = blockIdx.y * 128 + waveN * 64 + t * 16 + l15;
        const float bb = bias[n];
#pragma unroll
        for (int mt = 0; mt < 2; ++mt) {
            const int mbase = blockIdx.x * 128 + waveM * 32 + mt * 16 + half * 8;
#pragma unroll
            for (int r = 0; r < 8; ++r)
                out[(size_t)(mbase + r) * N + n] = c[mt][t][r] + bb;
        }
    }
}

// ---------------------------------------------------------------------------
// Stage 3: vsum[b,h,d] = sum_j V[b,j,h,d]   (V layout: [(b*L+j)*D + h*64 + d])
// ---------------------------------------------------------------------------
__global__ void vsum_kernel(const float* __restrict__ V, float* __restrict__ vsum) {
    const int b = blockIdx.x / H_, h = blockIdx.x % H_, d = threadIdx.x;
    const float* p = V + (size_t)b * L_ * D_ + h * DH_ + d;
    float s = 0.f;
    for (int j = 0; j < L_; ++j) s += p[(size_t)j * D_];
    vsum[blockIdx.x * DH_ + d] = s;
}

// ---------------------------------------------------------------------------
// Stage 4: log-sparse attention, one wave32 per (b,i,h) row.
//   out_i = (vsum + sum_{j in M(i)} (e_ij-1) v_j) / (L + sum (e_ij-1))
// Each lane owns head dims {2*lane, 2*lane+1}. Context written as bf16.
// ---------------------------------------------------------------------------
__global__ __launch_bounds__(256)
void logsparse_attn_kernel(const float* __restrict__ Q, const float* __restrict__ Km,
                           const float* __restrict__ V, const float* __restrict__ vsum,
                           unsigned short* __restrict__ ctx) {
    const int lane = threadIdx.x & 31;
    const int w = blockIdx.x * 8 + (threadIdx.x >> 5);
    const int h = w % H_;
    const int i = (w / H_) % L_;
    const int b = w / (H_ * L_);
    const size_t row = (size_t)(b * L_ + i) * D_ + h * DH_ + 2 * lane;
    const float2 q2 = *reinterpret_cast<const float2*>(Q + row);

    int js[16];
    int nj = 0;
    js[nj++] = i;                               // diagonal
    for (int k = 0; (1 << k) <= i + 1; ++k) {   // k = 0..floor(log2(i+1))
        const int j = i - (1 << k);
        if (j >= 0) js[nj++] = j;
    }

    float accx = 0.f, accy = 0.f, zacc = 0.f;
    for (int t = 0; t < nj; ++t) {
        const size_t kr = (size_t)(b * L_ + js[t]) * D_ + h * DH_ + 2 * lane;
        const float2 k2 = *reinterpret_cast<const float2*>(Km + kr);
        float p = q2.x * k2.x + q2.y * k2.y;
        p += __shfl_xor(p, 1, 32);              // wave32 butterfly reduce
        p += __shfl_xor(p, 2, 32);
        p += __shfl_xor(p, 4, 32);
        p += __shfl_xor(p, 8, 32);
        p += __shfl_xor(p, 16, 32);
        const float e = __expf(p * 0.125f) - 1.0f;   // scale = sqrt(64) = 8
        const float2 v2 = *reinterpret_cast<const float2*>(V + kr);
        accx += e * v2.x;
        accy += e * v2.y;
        zacc += e;
    }
    const float Z = (float)L_ + zacc;
    const float2 vs = *reinterpret_cast<const float2*>(vsum + (b * H_ + h) * DH_ + 2 * lane);
    const float ox = (vs.x + accx) / Z;
    const float oy = (vs.y + accy) / Z;
    const unsigned int packed = (unsigned int)f32_bf16_rne(ox) |
                                ((unsigned int)f32_bf16_rne(oy) << 16);
    *reinterpret_cast<unsigned int*>(ctx + row) = packed;
}

// ---------------------------------------------------------------------------
extern "C" void kernel_launch(void* const* d_in, const int* in_sizes, int n_in,
                              void* d_out, int out_size, void* d_ws, size_t ws_size,
                              hipStream_t stream) {
    const float* x  = (const float*)d_in[0];
    const float* Wq = (const float*)d_in[1];
    const float* bq = (const float*)d_in[2];
    const float* Wk = (const float*)d_in[3];
    const float* bk = (const float*)d_in[4];
    const float* Wv = (const float*)d_in[5];
    const float* bv = (const float*)d_in[6];
    const float* Wo = (const float*)d_in[7];
    const float* bo = (const float*)d_in[8];
    (void)in_sizes; (void)n_in; (void)out_size; (void)ws_size;

    const size_t M  = (size_t)B_ * L_;   // 4096 rows
    const size_t xN = M * D_;            // 4 Mi elements
    const size_t wN = (size_t)D_ * D_;   // 1 Mi elements

    char* base = (char*)d_ws;
    size_t off = 0;
    auto wsAlloc = [&](size_t bytes) -> void* {
        void* p = base + off;
        off = (off + bytes + 255) & ~(size_t)255;
        return p;
    };
    unsigned short* xb  = (unsigned short*)wsAlloc(xN * 2);
    unsigned short* Wqb = (unsigned short*)wsAlloc(wN * 2);
    unsigned short* Wkb = (unsigned short*)wsAlloc(wN * 2);
    unsigned short* Wvb = (unsigned short*)wsAlloc(wN * 2);
    unsigned short* Wob = (unsigned short*)wsAlloc(wN * 2);
    float* Qf = (float*)wsAlloc(xN * 4);
    float* Kf = (float*)wsAlloc(xN * 4);
    float* Vf = (float*)wsAlloc(xN * 4);
    float* vs = (float*)wsAlloc((size_t)B_ * H_ * DH_ * 4);
    unsigned short* ctx = (unsigned short*)wsAlloc(xN * 2);

    auto castTo = [&](const float* s, unsigned short* d, size_t n) {
        const int n4 = (int)(n / 4);
        cast_bf16_kernel<<<(n4 + 255) / 256, 256, 0, stream>>>(s, d, n4);
    };
    castTo(x,  xb,  xN);
    castTo(Wq, Wqb, wN);
    castTo(Wk, Wkb, wN);
    castTo(Wv, Wvb, wN);
    castTo(Wo, Wob, wN);

    dim3 grid((unsigned)(M / 128), D_ / 128);
    gemm_bf16_wmma<<<grid, 256, 0, stream>>>((const __bf16*)xb, (const __bf16*)Wqb, bq, Qf, (int)M, D_, D_);
    gemm_bf16_wmma<<<grid, 256, 0, stream>>>((const __bf16*)xb, (const __bf16*)Wkb, bk, Kf, (int)M, D_, D_);
    gemm_bf16_wmma<<<grid, 256, 0, stream>>>((const __bf16*)xb, (const __bf16*)Wvb, bv, Vf, (int)M, D_, D_);

    vsum_kernel<<<B_ * H_, DH_, 0, stream>>>(Vf, vs);
    logsparse_attn_kernel<<<(B_ * L_ * H_) / 8, 256, 0, stream>>>(Qf, Kf, Vf, vs, ctx);

    gemm_bf16_wmma<<<grid, 256, 0, stream>>>((const __bf16*)ctx, (const __bf16*)Wob, bo,
                                             (float*)d_out, (int)M, D_, D_);
}
